// BaseBlock_27814208209135
// MI455X (gfx1250) — matrix operand order repaired and checked by reading, not compiled, not run on hardware
//
#include <hip/hip_runtime.h>
#include <hip/hip_bf16.h>

// ---------------------------------------------------------------------------
// CDNA5 (gfx1250) transformer block: bf16 WMMA GEMMs with async global->LDS
// staging (ASYNCcnt path), double-buffered LDS, fp32 LN/softmax glue.
// ---------------------------------------------------------------------------

typedef __bf16 bf16x16 __attribute__((ext_vector_type(16)));
typedef float  f32x8   __attribute__((ext_vector_type(8)));
typedef unsigned short u16;

#define LDSP 40            // padded LDS row stride (halves): 80B rows, 16B-aligned
#define TILE_ELEMS (128 * LDSP)

__device__ __forceinline__ u16 f2bf(float f) {
    unsigned int u = __float_as_uint(f);
    unsigned int r = u + 0x7FFFu + ((u >> 16) & 1u);   // round-to-nearest-even
    return (u16)(r >> 16);
}
__device__ __forceinline__ float gelu_exact(float x) {
    return 0.5f * x * (1.0f + erff(x * 0.70710678118654752f));
}

// Async copy of 16 bytes global -> LDS (GLOBAL_LOAD_ASYNC_TO_LDS_B128, ASYNCcnt)
__device__ __forceinline__ void async_b128(unsigned lds_byte_off, const u16* g) {
    asm volatile("global_load_async_to_lds_b128 %0, %1, off"
                 :: "v"(lds_byte_off), "v"((unsigned long long)(uintptr_t)g)
                 : "memory");
}
__device__ __forceinline__ void wait_async0() {
#if __has_builtin(__builtin_amdgcn_s_wait_asynccnt)
    __builtin_amdgcn_s_wait_asynccnt(0);
#else
    asm volatile("s_wait_asynccnt 0x0" ::: "memory");
#endif
}

// A fragment: 16x32 bf16, M on lanes. lane l: row=l&15, kbase=(l>>4)*8
// halves 0..7 -> k=kbase..kbase+7 ; halves 8..15 -> k=16+kbase..16+kbase+7
__device__ __forceinline__ bf16x16 load_frag_a(const u16* lds, int row, int kbase) {
    union { uint4 q[2]; bf16x16 v; } u;
    u.q[0] = *(const uint4*)(lds + row * LDSP + kbase);
    u.q[1] = *(const uint4*)(lds + row * LDSP + 16 + kbase);
    return u.v;
}
// B fragment: 32x16 bf16, N on lanes; LDS tile n-major [n][k].
// lane l: n=l&15, kgrp=(l>>4)*8 ; halves 0..7 -> k=kgrp+i ; halves 8..15 -> k=16+kgrp+(i-8)
__device__ __forceinline__ bf16x16 load_frag_b(const u16* lds, int n, int kgrp) {
    union { uint4 q[2]; bf16x16 v; } u;
    u.q[0] = *(const uint4*)(lds + n * LDSP + kgrp);
    u.q[1] = *(const uint4*)(lds + n * LDSP + kgrp + 16);
    return u.v;
}

// ---------------------------------------------------------------------------
// bf16 WMMA GEMM:  C = act( scale * A @ Bt^T + bias ) (+ causal) (+ resid)
//   A  : [M][K] row-major bf16
//   Bt : [N][K] row-major bf16 (i.e. B pre-transposed; contiguous K)
// Block tile 128x128, BK=32, 8 waves, wave tile 32x64, double-buffered async LDS.
// Requires M%128==0, N%128==0, K%32==0 (holds for every GEMM here).
// ---------------------------------------------------------------------------
__global__ __launch_bounds__(256)
void gemm_bf16_wmma(const u16* __restrict__ A, long long strideA, int lda,
                    const u16* __restrict__ Bt, long long strideB, int ldb,
                    const float* __restrict__ bias,
                    const float* __restrict__ resid, long long strideR,
                    float* __restrict__ outF, u16* __restrict__ outB,
                    long long strideC, int outBT,
                    int M, int N, int K,
                    float scale, int act, int causal)
{
    __shared__ u16 ldsbuf[2][2][TILE_ELEMS];   // [buffer][A/B][tile] = 40 KB

    const int tid  = threadIdx.x;
    const int lane = tid & 31;
    const int wave = tid >> 5;          // 0..7
    const int wm   = wave >> 1;         // 0..3 -> 32-row strip
    const int wn   = wave & 1;          // 0..1 -> 64-col strip
    const int m0   = blockIdx.y * 128;
    const int n0   = blockIdx.x * 128;
    const int z    = blockIdx.z;

    A  += (long long)z * strideA;
    Bt += (long long)z * strideB;
    if (outF)  outF  += (long long)z * strideC;
    if (outB)  outB  += (long long)z * strideC;
    if (resid) resid += (long long)z * strideR;

    // Whole tile strictly above the causal diagonal: emit -inf and exit.
    if (causal && n0 > m0 + 127) {
        if (outF) {
            #pragma unroll 4
            for (int i = 0; i < 64; ++i) {
                int idx = i * 256 + tid;
                int m = idx >> 7, n = idx & 127;
                outF[(size_t)(m0 + m) * N + (n0 + n)] = -__builtin_inff();
            }
        }
        return;
    }

    // LDS byte offsets of the four tiles (low 32 bits of flat addr = LDS offset)
    const unsigned aOff[2] = { (unsigned)(uintptr_t)&ldsbuf[0][0][0],
                               (unsigned)(uintptr_t)&ldsbuf[1][0][0] };
    const unsigned bOff[2] = { (unsigned)(uintptr_t)&ldsbuf[0][1][0],
                               (unsigned)(uintptr_t)&ldsbuf[1][1][0] };

    // Per-thread staging chunks: 512 b128 chunks per 128x32 tile, 2 per thread.
    const int c0 = tid,        r0 = c0 >> 2, k0c = (c0 & 3) * 8;
    const int c1 = 256 + tid,  r1 = c1 >> 2, k1c = (c1 & 3) * 8;
    const u16* Ab = A  + (size_t)m0 * lda;
    const u16* Bb = Bt + (size_t)n0 * ldb;

    #define STAGE(buf, kk)                                                      \
        do {                                                                    \
            async_b128(aOff[buf] + (unsigned)(r0 * LDSP + k0c) * 2,             \
                       Ab + (size_t)r0 * lda + (kk) + k0c);                     \
            async_b128(aOff[buf] + (unsigned)(r1 * LDSP + k1c) * 2,             \
                       Ab + (size_t)r1 * lda + (kk) + k1c);                     \
            async_b128(bOff[buf] + (unsigned)(r0 * LDSP + k0c) * 2,             \
                       Bb + (size_t)r0 * ldb + (kk) + k0c);                     \
            async_b128(bOff[buf] + (unsigned)(r1 * LDSP + k1c) * 2,             \
                       Bb + (size_t)r1 * ldb + (kk) + k1c);                     \
        } while (0)

    f32x8 acc[2][4] = {};

    const int rA    = lane & 15;
    const int kbase = (lane >> 4) * 8;
    const int kgrp  = (lane >> 4) * 16;
    const int nk    = K >> 5;

    STAGE(0, 0);
    wait_async0();
    __syncthreads();

    for (int it = 0; it < nk; ++it) {
        const int buf = it & 1;
        if (it + 1 < nk) STAGE(buf ^ 1, (it + 1) << 5);   // overlap with WMMA

        const u16* lA = &ldsbuf[buf][0][0];
        const u16* lB = &ldsbuf[buf][1][0];

        bf16x16 afrag[2], bfrag[4];
        #pragma unroll
        for (int ms = 0; ms < 2; ++ms)
            afrag[ms] = load_frag_a(lA, wm * 32 + ms * 16 + rA, kbase);
        #pragma unroll
        for (int ns = 0; ns < 4; ++ns)
            bfrag[ns] = load_frag_b(lB, wn * 64 + ns * 16 + rA, kbase);

        #pragma unroll
        for (int ms = 0; ms < 2; ++ms)
            #pragma unroll
            for (int ns = 0; ns < 4; ++ns)
                acc[ms][ns] = __builtin_amdgcn_wmma_f32_16x16x32_bf16(
                    false, afrag[ms], false, bfrag[ns],
                    (short)0, acc[ms][ns], false, false);

        wait_async0();       // my next-buffer tiles have landed
        __syncthreads();     // everyone's tiles visible; prev buffer reusable
    }
    #undef STAGE

    // ---- epilogue: C layout lane l, vgpr r -> M = r + 8*(l>>4), N = l&15 ----
    const int mBase  = m0 + wm * 32;
    const int nBase  = n0 + wn * 64;
    const int rowOff = (lane >> 4) * 8;
    const int colOff = lane & 15;
    #pragma unroll
    for (int ms = 0; ms < 2; ++ms) {
        #pragma unroll
        for (int ns = 0; ns < 4; ++ns) {
            #pragma unroll
            for (int r = 0; r < 8; ++r) {
                int m = mBase + ms * 16 + rowOff + r;
                int n = nBase + ns * 16 + colOff;
                float v = acc[ms][ns][r] * scale;
                if (bias)   v += bias[n];
                if (act)    v  = gelu_exact(v);
                if (causal && n > m) v = -__builtin_inff();
                size_t o = (size_t)m * N + n;
                if (resid)  v += resid[o];
                if (outF)   outF[o] = v;
                if (outB) {
                    size_t ob = outBT ? ((size_t)n * M + m) : o;
                    outB[ob] = f2bf(v);
                }
            }
        }
    }
}

// ---------------------------------------------------------------------------
// LayerNorm over rows of 1024 fp32 -> bf16
// ---------------------------------------------------------------------------
__global__ __launch_bounds__(256)
void ln_to_bf16(const float* __restrict__ x, const float* __restrict__ w,
                const float* __restrict__ b, u16* __restrict__ out)
{
    __shared__ float red[256];
    const int tid = threadIdx.x;
    const size_t row = blockIdx.x;
    const float* xr = x + row * 1024;

    float lx[4]; float s = 0.f;
    #pragma unroll
    for (int i = 0; i < 4; ++i) { lx[i] = xr[tid + i * 256]; s += lx[i]; }
    red[tid] = s; __syncthreads();
    for (int off = 128; off > 0; off >>= 1) {
        if (tid < off) red[tid] += red[tid + off];
        __syncthreads();
    }
    const float mu = red[0] * (1.0f / 1024.0f);
    __syncthreads();

    float vs = 0.f;
    #pragma unroll
    for (int i = 0; i < 4; ++i) { float d = lx[i] - mu; vs += d * d; }
    red[tid] = vs; __syncthreads();
    for (int off = 128; off > 0; off >>= 1) {
        if (tid < off) red[tid] += red[tid + off];
        __syncthreads();
    }
    const float rstd = rsqrtf(red[0] * (1.0f / 1024.0f) + 1e-5f);

    #pragma unroll
    for (int i = 0; i < 4; ++i) {
        int c = tid + i * 256;
        out[row * 1024 + c] = f2bf((lx[i] - mu) * rstd * w[c] + b[c]);
    }
}

// ---------------------------------------------------------------------------
// Row softmax over 2048 fp32 logits (with -inf causal mask) -> bf16 probs
// ---------------------------------------------------------------------------
__global__ __launch_bounds__(256)
void softmax_to_bf16(const float* __restrict__ S, u16* __restrict__ P)
{
    __shared__ float red[256];
    const int tid = threadIdx.x;
    const size_t row = blockIdx.x;
    const float* sr = S + row * 2048;

    float lv[8]; float mx = -__builtin_inff();
    #pragma unroll
    for (int i = 0; i < 8; ++i) { lv[i] = sr[tid + i * 256]; mx = fmaxf(mx, lv[i]); }
    red[tid] = mx; __syncthreads();
    for (int off = 128; off > 0; off >>= 1) {
        if (tid < off) red[tid] = fmaxf(red[tid], red[tid + off]);
        __syncthreads();
    }
    mx = red[0]; __syncthreads();

    float sum = 0.f;
    #pragma unroll
    for (int i = 0; i < 8; ++i) { float e = __expf(lv[i] - mx); lv[i] = e; sum += e; }
    red[tid] = sum; __syncthreads();
    for (int off = 128; off > 0; off >>= 1) {
        if (tid < off) red[tid] += red[tid + off];
        __syncthreads();
    }
    const float inv = 1.0f / red[0];

    #pragma unroll
    for (int i = 0; i < 8; ++i)
        P[row * 2048 + tid + i * 256] = f2bf(lv[i] * inv);
}

// ---------------------------------------------------------------------------
// fp32 [K][N] -> bf16 transposed [N][K], LDS-tiled (32x32 tiles)
// ---------------------------------------------------------------------------
__global__ __launch_bounds__(256)
void cast_transpose_f32_bf16(const float* __restrict__ src, u16* __restrict__ dst,
                             int K, int N)
{
    __shared__ u16 t[32][33];
    const int bk = blockIdx.y * 32, bn = blockIdx.x * 32;
    const int tx = threadIdx.x & 31, ty = threadIdx.x >> 5;   // 32 x 8
    #pragma unroll
    for (int i = 0; i < 4; ++i) {
        int k = ty + i * 8;
        t[k][tx] = f2bf(src[(size_t)(bk + k) * N + bn + tx]);
    }
    __syncthreads();
    #pragma unroll
    for (int i = 0; i < 4; ++i) {
        int n = ty + i * 8;
        dst[(size_t)(bn + n) * K + bk + tx] = t[tx][n];
    }
}

// ---------------------------------------------------------------------------
// Host-side orchestration
// ---------------------------------------------------------------------------
extern "C" void kernel_launch(void* const* d_in, const int* in_sizes, int n_in,
                              void* d_out, int out_size, void* d_ws, size_t ws_size,
                              hipStream_t stream)
{
    (void)in_sizes; (void)n_in; (void)out_size; (void)ws_size;
    constexpr int E = 1024, H = 4096, B = 4, T = 2048;
    constexpr long long TE = (long long)T * E;
    constexpr long long TT = (long long)T * T;
    const int M = B * T;                           // 8192 rows

    const float* x     = (const float*)d_in[0];
    const float* ln1_w = (const float*)d_in[1];
    const float* ln1_b = (const float*)d_in[2];
    const float* Wk    = (const float*)d_in[3];
    const float* bk    = (const float*)d_in[4];
    const float* Wq    = (const float*)d_in[5];
    const float* bq    = (const float*)d_in[6];
    const float* Wv    = (const float*)d_in[7];
    const float* bv    = (const float*)d_in[8];
    const float* ln2_w = (const float*)d_in[9];
    const float* ln2_b = (const float*)d_in[10];
    const float* W1    = (const float*)d_in[11];
    const float* b1    = (const float*)d_in[12];
    const float* W2a   = (const float*)d_in[13];
    const float* b2a   = (const float*)d_in[14];
    const float* W2b   = (const float*)d_in[15];
    const float* b2b   = (const float*)d_in[16];
    const float* W3    = (const float*)d_in[17];
    const float* b3    = (const float*)d_in[18];
    float* out = (float*)d_out;

    // Workspace layout (MiB offsets), overlapped across phases.
    // All bf16 weight copies are stored TRANSPOSED: [N][K].
    char* ws = (char*)d_ws;
    const size_t MiB = 1ull << 20;
    u16* wWkT  = (u16*)(ws + 0   * MiB);  // 2 MiB  [E][E]
    u16* wWqT  = (u16*)(ws + 2   * MiB);  // 2
    u16* wWvT  = (u16*)(ws + 4   * MiB);  // 2
    u16* wW1T  = (u16*)(ws + 6   * MiB);  // 8      [H][E]
    u16* wW2aT = (u16*)(ws + 14  * MiB);  // 32     [H][H]
    u16* wW2bT = (u16*)(ws + 46  * MiB);  // 32
    u16* wW3T  = (u16*)(ws + 78  * MiB);  // 8      [E][H]
    u16* hln   = (u16*)(ws + 86  * MiB);  // 16 (later part of f2 region)
    u16* kbuf  = (u16*)(ws + 102 * MiB);  // 16
    u16* qbuf  = (u16*)(ws + 118 * MiB);  // 16
    u16* vT    = (u16*)(ws + 134 * MiB);  // 16  v stored transposed [E][B*T]
    float* Sb  = (float*)(ws + 150 * MiB);// 64 fp32 (later f1 / f3)
    u16* Pb    = (u16*)(ws + 214 * MiB);  // 32 (later h2)
    u16* h2    = (u16*)(ws + 214 * MiB);  // reuses P (dead after att GEMM)
    u16* f1    = (u16*)(ws + 150 * MiB);  // reuses S (dead after softmax)
    u16* f2    = (u16*)(ws + 86  * MiB);  // reuses hln/k/q/v (dead)
    u16* f3    = (u16*)(ws + 150 * MiB);  // reuses f1 (dead after f2 GEMM)

    // 1) cast + transpose all weights fp32 -> bf16 [N][K] (L2-resident later)
    cast_transpose_f32_bf16<<<dim3(E / 32, E / 32), 256, 0, stream>>>(Wk,  wWkT,  E, E);
    cast_transpose_f32_bf16<<<dim3(E / 32, E / 32), 256, 0, stream>>>(Wq,  wWqT,  E, E);
    cast_transpose_f32_bf16<<<dim3(E / 32, E / 32), 256, 0, stream>>>(Wv,  wWvT,  E, E);
    cast_transpose_f32_bf16<<<dim3(H / 32, E / 32), 256, 0, stream>>>(W1,  wW1T,  E, H);
    cast_transpose_f32_bf16<<<dim3(H / 32, H / 32), 256, 0, stream>>>(W2a, wW2aT, H, H);
    cast_transpose_f32_bf16<<<dim3(H / 32, H / 32), 256, 0, stream>>>(W2b, wW2bT, H, H);
    cast_transpose_f32_bf16<<<dim3(E / 32, H / 32), 256, 0, stream>>>(W3,  wW3T,  H, E);

    // 2) LN1
    ln_to_bf16<<<M, 256, 0, stream>>>(x, ln1_w, ln1_b, hln);

    // 3) k / q / v projections   [8192,1024] x [1024,1024]; v written transposed
    {
        dim3 g(E / 128, M / 128, 1);
        gemm_bf16_wmma<<<g, 256, 0, stream>>>(hln, 0, E, wWkT, 0, E, bk,
            nullptr, 0, nullptr, kbuf, 0, 0, M, E, E, 1.0f, 0, 0);
        gemm_bf16_wmma<<<g, 256, 0, stream>>>(hln, 0, E, wWqT, 0, E, bq,
            nullptr, 0, nullptr, qbuf, 0, 0, M, E, E, 1.0f, 0, 0);
        gemm_bf16_wmma<<<g, 256, 0, stream>>>(hln, 0, E, wWvT, 0, E, bv,
            nullptr, 0, nullptr, vT,   0, 1, M, E, E, 1.0f, 0, 0);
    }

    // 4) S = k @ q^T / sqrt(E), causal  (q is already [N][K]; grid.z = batch)
    {
        dim3 g(T / 128, T / 128, B);
        gemm_bf16_wmma<<<g, 256, 0, stream>>>(kbuf, TE, E, qbuf, TE, E, nullptr,
            nullptr, 0, Sb, nullptr, TT, 0, T, T, E, 0.03125f, 0, 1);
    }

    // 5) row softmax -> bf16 probabilities
    softmax_to_bf16<<<B * T, 256, 0, stream>>>(Sb, Pb);

    // 6) x1 = x + P @ v  (B = v^T [E][8192]: ldb=M, batch offset = b*T columns)
    {
        dim3 g(E / 128, T / 128, B);
        gemm_bf16_wmma<<<g, 256, 0, stream>>>(Pb, TT, T, vT, (long long)T, M, nullptr,
            x, TE, out, nullptr, TE, 0, T, E, T, 1.0f, 0, 0);
    }

    // 7) LN2 on x1
    ln_to_bf16<<<M, 256, 0, stream>>>(out, ln2_w, ln2_b, h2);

    // 8) f1 = gelu(h2 @ W1 + b1)
    {
        dim3 g(H / 128, M / 128, 1);
        gemm_bf16_wmma<<<g, 256, 0, stream>>>(h2, 0, E, wW1T, 0, E, b1,
            nullptr, 0, nullptr, f1, 0, 0, M, H, E, 1.0f, 1, 0);
    }
    // 9) f2 = gelu(f1 @ W2a + b2a)
    {
        dim3 g(H / 128, M / 128, 1);
        gemm_bf16_wmma<<<g, 256, 0, stream>>>(f1, 0, H, wW2aT, 0, H, b2a,
            nullptr, 0, nullptr, f2, 0, 0, M, H, H, 1.0f, 1, 0);
    }
    // 10) f3 = gelu(f2 @ W2b + b2b)
    {
        dim3 g(H / 128, M / 128, 1);
        gemm_bf16_wmma<<<g, 256, 0, stream>>>(f2, 0, H, wW2bT, 0, H, b2b,
            nullptr, 0, nullptr, f3, 0, 0, M, H, H, 1.0f, 1, 0);
    }
    // 11) out = x1 + f3 @ W3 + b3   (reads+overwrites d_out 1:1, safe)
    {
        dim3 g(E / 128, M / 128, 1);
        gemm_bf16_wmma<<<g, 256, 0, stream>>>(f3, 0, H, wW3T, 0, H, b3,
            out, TE, out, nullptr, TE, 0, M, E, H, 1.0f, 0, 0);
    }
}
